// EdgeAwareGraphTransformerLayer_7696581394498
// MI455X (gfx1250) — matrix operand order repaired and checked by reading, not compiled, not run on hardware
//
#include <hip/hip_runtime.h>
#include <hip/hip_bf16.h>
#include <math.h>

typedef unsigned short u16;
typedef __attribute__((ext_vector_type(16))) __bf16 bf16x16;
typedef __attribute__((ext_vector_type(8)))  __bf16 bf16x8;
typedef __attribute__((ext_vector_type(8)))  float  f32x8;

#define G_   64
#define N_   256
#define H_   8
#define D_   32
#define INDIM 256
#define FFDIM 1024
#define ROWS (G_ * N_)          // 16384
#define E_   (G_ * N_ * 16)     // 262144

__device__ __forceinline__ u16 f2bf(float f) {
  unsigned u = __float_as_uint(f);
  unsigned r = 0x7FFFu + ((u >> 16) & 1u);
  return (u16)((u + r) >> 16);
}

__device__ __forceinline__ bf16x16 load_frag(const u16* p0, const u16* p1) {
  union { bf16x16 v; bf16x8 h[2]; } u;
  u.h[0] = *(const bf16x8*)p0;
  u.h[1] = *(const bf16x8*)p1;
  return u.v;
}

__device__ __forceinline__ f32x8 wmma_bf16(bf16x16 a, bf16x16 b, f32x8 c) {
  return __builtin_amdgcn_wmma_f32_16x16x32_bf16(false, a, false, b, (short)0, c,
                                                 false, false);
}

// CDNA5 async copy: 16B global -> LDS per lane, tracked by ASYNCcnt.
// lds_off = wave-relative LDS byte address (low 32 bits of generic pointer).
__device__ __forceinline__ void async_g2l_b128(const void* lds_ptr, const void* gptr) {
  unsigned lds_off = (unsigned)(size_t)lds_ptr;
  asm volatile("global_load_async_to_lds_b128 %0, %1, off"
               :: "v"(lds_off), "v"(gptr) : "memory");
}
__device__ __forceinline__ void wait_asynccnt0() {
  asm volatile("s_wait_asynccnt 0x0" ::: "memory");
}

// ---------------------------------------------------------------- utilities
__global__ void zero_f32_kernel(float* __restrict__ p, int n) {
  int i = blockIdx.x * 256 + threadIdx.x;
  if (i < n) p[i] = 0.0f;
}

__global__ void cast_bf16_kernel(const float* __restrict__ in, u16* __restrict__ out, int n) {
  int i = blockIdx.x * 256 + threadIdx.x;
  if (i < n) out[i] = f2bf(in[i]);
}

// LayerNorm (one wave per row) + cast to bf16
__global__ __launch_bounds__(256)
void ln_cast_kernel(const float* __restrict__ in, const float* __restrict__ w,
                    const float* __restrict__ b, u16* __restrict__ out) {
  const int cols = INDIM;
  int row  = blockIdx.x * 8 + (threadIdx.x >> 5);
  int lane = threadIdx.x & 31;
  const float* r = in + (size_t)row * cols;
  float s = 0.f, s2 = 0.f;
  for (int j = lane; j < cols; j += 32) { float x = r[j]; s += x; s2 += x * x; }
  for (int off = 16; off > 0; off >>= 1) {
    s  += __shfl_xor(s,  off, 32);
    s2 += __shfl_xor(s2, off, 32);
  }
  float m  = s / cols;
  float var = s2 / cols - m * m;
  float rs = rsqrtf(var + 1e-5f);
  u16* o = out + (size_t)row * cols;
  for (int j = lane; j < cols; j += 32)
    o[j] = f2bf((r[j] - m) * rs * w[j] + b[j]);
}

// Edge bias: ew = edge_attr @ We + be, scattered (atomic) into [G,H,N,N]
__global__ void edge_bias_kernel(const long long* __restrict__ ei,
                                 const float* __restrict__ ea,
                                 const float* __restrict__ We,
                                 const float* __restrict__ be,
                                 float* __restrict__ ebias) {
  int e = blockIdx.x * 256 + threadIdx.x;
  if (e >= E_) return;
  long long s = ei[e], d = ei[E_ + e];
  int g  = (int)(s >> 8);
  int ls = (int)(s & 255);
  int ld = (int)(d & 255);
  float a0 = ea[e * 2], a1 = ea[e * 2 + 1];
#pragma unroll
  for (int h = 0; h < H_; ++h) {
    float ew = a0 * We[h] + a1 * We[H_ + h] + be[h];
    atomicAdd(&ebias[(((size_t)g * H_ + h) * N_ + ls) * N_ + ld], ew);
  }
}

// ------------------------------------------------- generic bf16 WMMA GEMM
// C[M,N] = act(A[M,K] @ W[K,N] + bias[N]) (+ resid). Compile-time epilogue.
// Block: 256 threads = 8 waves; tile 128x128; K step 32.
// A tile staged via global_load_async_to_lds_b128 (ASYNCcnt); W tile
// transposed through VGPRs (2-byte scatter cannot use the async path).
template <bool RELU, bool RESID, bool OUTF>
__global__ __launch_bounds__(256)
void gemm_bf16_wmma(const u16* __restrict__ A, const u16* __restrict__ W,
                    const float* __restrict__ bias,
                    u16* __restrict__ outB, float* __restrict__ outF,
                    const float* __restrict__ resid,
                    int M, int N, int K) {
  __shared__ u16 Asl[128 * 32];   // [row][k]
  __shared__ u16 Btl[128 * 32];   // [n][k]  (transposed W tile)
  int tid  = threadIdx.x;
  int lane = tid & 31;
  int w    = tid >> 5;
  int mBase = blockIdx.x * 128;
  int nBase = blockIdx.y * 128;
  int wm = (w >> 2) * 64;          // 0 / 64
  int wn = (w & 3) * 32;           // 0..96

  f32x8 acc[4][2];
#pragma unroll
  for (int mf = 0; mf < 4; ++mf)
#pragma unroll
    for (int nf = 0; nf < 2; ++nf)
#pragma unroll
      for (int i = 0; i < 8; ++i) acc[mf][nf][i] = 0.f;

  for (int k0 = 0; k0 < K; k0 += 32) {
    // stage A tile via async DMA to LDS (512 chunks of 16B)
#pragma unroll
    for (int i = 0; i < 2; ++i) {
      int c = tid * 2 + i;
      int r = c >> 2;
      int kk = (c & 3) * 8;
      async_g2l_b128(&Asl[r * 32 + kk], &A[(size_t)(mBase + r) * K + k0 + kk]);
      if (k0 + 32 < K)  // hint L2 for next K-step (global_prefetch_b8)
        __builtin_prefetch(&A[(size_t)(mBase + r) * K + k0 + 32 + kk], 0, 1);
    }
    // stage W tile transposed -> Btl[n][k]
#pragma unroll
    for (int i = 0; i < 2; ++i) {
      int c = tid * 2 + i;
      int kk = c >> 4;                 // 0..31
      int nn = (c & 15) * 8;
      uint4 t = *(const uint4*)&W[(size_t)(k0 + kk) * N + nBase + nn];
      u16 tmp[8]; *(uint4*)tmp = t;
#pragma unroll
      for (int j = 0; j < 8; ++j) Btl[(nn + j) * 32 + kk] = tmp[j];
    }
    wait_asynccnt0();
    __syncthreads();

    bf16x16 af[4], bfr[2];
#pragma unroll
    for (int mf = 0; mf < 4; ++mf) {
      int row  = wm + mf * 16 + (lane & 15);
      int koff = (lane < 16) ? 0 : 8;
      af[mf] = load_frag(&Asl[row * 32 + koff], &Asl[row * 32 + koff + 16]);
    }
#pragma unroll
    for (int nf = 0; nf < 2; ++nf) {
      int col  = wn + nf * 16 + (lane & 15);
      int koff = (lane < 16) ? 0 : 8;
      bfr[nf] = load_frag(&Btl[col * 32 + koff], &Btl[col * 32 + koff + 16]);
    }
#pragma unroll
    for (int mf = 0; mf < 4; ++mf)
#pragma unroll
      for (int nf = 0; nf < 2; ++nf)
        acc[mf][nf] = wmma_bf16(af[mf], bfr[nf], acc[mf][nf]);
    __syncthreads();
  }

  int rowb = mBase + wm + ((lane >= 16) ? 8 : 0);
  int colb = nBase + wn + (lane & 15);
#pragma unroll
  for (int mf = 0; mf < 4; ++mf)
#pragma unroll
    for (int nf = 0; nf < 2; ++nf) {
      float bval = bias[colb + nf * 16];
#pragma unroll
      for (int i = 0; i < 8; ++i) {
        int row = rowb + mf * 16 + i;
        int col = colb + nf * 16;
        float v = acc[mf][nf][i] + bval;
        if (RELU) v = fmaxf(v, 0.f);
        size_t idx = (size_t)row * N + col;
        if (RESID) v += resid[idx];
        if (OUTF) outF[idx] = v;
        else      outB[idx] = f2bf(v);
      }
    }
}

// ------------------------------------------------- fused attention per (g,h)
// scores = q k^T / sqrt(D) + ebias; softmax; out = p v; xout = x + out
// LDS: k[256x32]bf16 | vT[32x256]bf16 | sc[8][16x256]f32 | p[8][16x256]bf16
__global__ __launch_bounds__(256)
void attn_kernel(const u16* __restrict__ q, const u16* __restrict__ k,
                 const u16* __restrict__ v, const float* __restrict__ ebias,
                 const float* __restrict__ x, float* __restrict__ xout) {
  extern __shared__ char smem[];
  u16*   kl = (u16*)smem;                               // 16 KB
  u16*   vt = (u16*)(smem + 16 * 1024);                 // 16 KB (vt[d*256+key])
  float* sc = (float*)(smem + 32 * 1024);               // 128 KB
  u16*   pl = (u16*)(smem + 32 * 1024 + 128 * 1024);    // 64 KB

  int g = blockIdx.x >> 3;
  int h = blockIdx.x & 7;
  int tid = threadIdx.x, lane = tid & 31, w = tid >> 5;
  size_t base = ((size_t)g * N_) * INDIM + (size_t)h * D_;

  // stage K via async DMA, V^T via VGPR transpose
#pragma unroll
  for (int i = 0; i < 4; ++i) {
    int c = tid * 4 + i;                 // 1024 chunks
    int n = c >> 2;
    int doff = (c & 3) * 8;
    async_g2l_b128(&kl[n * 32 + doff], &k[base + (size_t)n * INDIM + doff]);
    uint4 tv = *(const uint4*)&v[base + (size_t)n * INDIM + doff];
    u16 tmp[8]; *(uint4*)tmp = tv;
#pragma unroll
    for (int j = 0; j < 8; ++j) vt[(doff + j) * N_ + n] = tmp[j];
  }
  wait_asynccnt0();
  __syncthreads();

  float* mysc = sc + w * 16 * N_;
  u16*   myp  = pl + w * 16 * N_;
  const float scale = 0.17677669529663687f;   // 1/sqrt(32)
  const float* eb = &ebias[((size_t)(g * H_ + h)) * N_ * N_];

  for (int rb = 0; rb < 2; ++rb) {
    int mb = w * 32 + rb * 16;
    // Q fragment (direct from global; L2-resident)
    int qrow = mb + (lane & 15);
    int koff = (lane < 16) ? 0 : 8;
    bf16x16 qf = load_frag(&q[base + (size_t)qrow * INDIM + koff],
                           &q[base + (size_t)qrow * INDIM + koff + 16]);
    // scores: 16 key-tiles, one WMMA each (D == 32 == K-step)
    for (int nt = 0; nt < 16; ++nt) {
      int kcol = nt * 16 + (lane & 15);
      int kk = (lane < 16) ? 0 : 8;
      bf16x16 bf = load_frag(&kl[kcol * 32 + kk], &kl[kcol * 32 + kk + 16]);
      f32x8 c;
#pragma unroll
      for (int i = 0; i < 8; ++i) c[i] = 0.f;
      c = wmma_bf16(qf, bf, c);
      int col = nt * 16 + (lane & 15);
#pragma unroll
      for (int i = 0; i < 8; ++i) {
        int row = mb + i + ((lane >= 16) ? 8 : 0);
        mysc[(row - mb) * N_ + col] = c[i] * scale + eb[(size_t)row * N_ + col];
      }
    }
    __syncthreads();
    // softmax: 2 lanes per row, 128 cols each
    {
      int row = lane & 15, half = lane >> 4;
      float* rp = mysc + row * N_ + half * 128;
      float mx = -3.4e38f;
      for (int j = 0; j < 128; ++j) mx = fmaxf(mx, rp[j]);
      mx = fmaxf(mx, __shfl_xor(mx, 16, 32));
      float sum = 0.f;
      for (int j = 0; j < 128; ++j) { float e = __expf(rp[j] - mx); rp[j] = e; sum += e; }
      sum += __shfl_xor(sum, 16, 32);
      float inv = 1.f / sum;
      u16* pr = myp + row * N_ + half * 128;
      for (int j = 0; j < 128; ++j) pr[j] = f2bf(rp[j] * inv);
    }
    __syncthreads();
    // out = p @ v : 2 d-tiles, K loop of 8x32
#pragma unroll
    for (int df = 0; df < 2; ++df) {
      f32x8 acc;
#pragma unroll
      for (int i = 0; i < 8; ++i) acc[i] = 0.f;
      for (int kk = 0; kk < 8; ++kk) {
        int m  = lane & 15;
        int ko = (lane < 16) ? 0 : 8;
        bf16x16 pf = load_frag(&myp[m * N_ + kk * 32 + ko],
                               &myp[m * N_ + kk * 32 + ko + 16]);
        int d   = df * 16 + (lane & 15);
        int ko2 = (lane < 16) ? 0 : 8;
        bf16x16 vf = load_frag(&vt[d * N_ + kk * 32 + ko2],
                               &vt[d * N_ + kk * 32 + ko2 + 16]);
        acc = wmma_bf16(pf, vf, acc);
      }
#pragma unroll
      for (int i = 0; i < 8; ++i) {
        int row = mb + i + ((lane >= 16) ? 8 : 0);
        int d = df * 16 + (lane & 15);
        size_t idx = base + (size_t)row * INDIM + d;
        xout[idx] = x[idx] + acc[i];
      }
    }
  }
}

// ---------------------------------------------------------------- launcher
extern "C" void kernel_launch(void* const* d_in, const int* in_sizes, int n_in,
                              void* d_out, int out_size, void* d_ws, size_t ws_size,
                              hipStream_t stream) {
  const float*     x    = (const float*)d_in[0];
  const long long* ei   = (const long long*)d_in[1];
  const float*     ea   = (const float*)d_in[2];
  const float*     ln1w = (const float*)d_in[4];
  const float*     ln1b = (const float*)d_in[5];
  const float*     Wq   = (const float*)d_in[6];
  const float*     bq   = (const float*)d_in[7];
  const float*     Wk   = (const float*)d_in[8];
  const float*     bk   = (const float*)d_in[9];
  const float*     Wv   = (const float*)d_in[10];
  const float*     bv   = (const float*)d_in[11];
  const float*     We   = (const float*)d_in[12];
  const float*     be   = (const float*)d_in[13];
  const float*     ln2w = (const float*)d_in[14];
  const float*     ln2b = (const float*)d_in[15];
  const float*     Wf1  = (const float*)d_in[16];
  const float*     bf1  = (const float*)d_in[17];
  const float*     Wf2  = (const float*)d_in[18];
  const float*     bf2  = (const float*)d_in[19];

  char* ws = (char*)d_ws;
  size_t off = 0;
  auto alloc = [&](size_t bytes) -> void* {
    void* p = ws + off;
    off += (bytes + 255) & ~(size_t)255;
    return p;
  };
  u16*   xn   = (u16*)alloc((size_t)ROWS * INDIM * 2);
  u16*   wqb  = (u16*)alloc((size_t)INDIM * INDIM * 2);
  u16*   wkb  = (u16*)alloc((size_t)INDIM * INDIM * 2);
  u16*   wvb  = (u16*)alloc((size_t)INDIM * INDIM * 2);
  u16*   wf1b = (u16*)alloc((size_t)INDIM * FFDIM * 2);
  u16*   wf2b = (u16*)alloc((size_t)FFDIM * INDIM * 2);
  u16*   qb   = (u16*)alloc((size_t)ROWS * INDIM * 2);
  u16*   kb   = (u16*)alloc((size_t)ROWS * INDIM * 2);
  u16*   vb   = (u16*)alloc((size_t)ROWS * INDIM * 2);
  float* eb   = (float*)alloc((size_t)G_ * H_ * N_ * N_ * 4);
  float* xo   = (float*)alloc((size_t)ROWS * INDIM * 4);
  u16*   hb   = (u16*)alloc((size_t)ROWS * INDIM * 2);
  u16*   ff1  = (u16*)alloc((size_t)ROWS * FFDIM * 2);

  const int ebN = G_ * H_ * N_ * N_;
  zero_f32_kernel<<<(ebN + 255) / 256, 256, 0, stream>>>(eb, ebN);
  cast_bf16_kernel<<<(INDIM * INDIM + 255) / 256, 256, 0, stream>>>(Wq, wqb, INDIM * INDIM);
  cast_bf16_kernel<<<(INDIM * INDIM + 255) / 256, 256, 0, stream>>>(Wk, wkb, INDIM * INDIM);
  cast_bf16_kernel<<<(INDIM * INDIM + 255) / 256, 256, 0, stream>>>(Wv, wvb, INDIM * INDIM);
  cast_bf16_kernel<<<(INDIM * FFDIM + 255) / 256, 256, 0, stream>>>(Wf1, wf1b, INDIM * FFDIM);
  cast_bf16_kernel<<<(FFDIM * INDIM + 255) / 256, 256, 0, stream>>>(Wf2, wf2b, FFDIM * INDIM);

  ln_cast_kernel<<<ROWS / 8, 256, 0, stream>>>(x, ln1w, ln1b, xn);

  dim3 gQ(ROWS / 128, INDIM / 128);
  gemm_bf16_wmma<false, false, false><<<gQ, 256, 0, stream>>>(
      xn, wqb, bq, qb, nullptr, nullptr, ROWS, INDIM, INDIM);
  gemm_bf16_wmma<false, false, false><<<gQ, 256, 0, stream>>>(
      xn, wkb, bk, kb, nullptr, nullptr, ROWS, INDIM, INDIM);
  gemm_bf16_wmma<false, false, false><<<gQ, 256, 0, stream>>>(
      xn, wvb, bv, vb, nullptr, nullptr, ROWS, INDIM, INDIM);

  edge_bias_kernel<<<(E_ + 255) / 256, 256, 0, stream>>>(ei, ea, We, be, eb);

  size_t attn_lds = 16 * 1024 + 16 * 1024 + 128 * 1024 + 64 * 1024;  // 224 KB
  attn_kernel<<<G_ * H_, 256, attn_lds, stream>>>(qb, kb, vb, eb, x, xo);

  ln_cast_kernel<<<ROWS / 8, 256, 0, stream>>>(xo, ln2w, ln2b, hb);

  dim3 gF1(ROWS / 128, FFDIM / 128);
  gemm_bf16_wmma<true, false, false><<<gF1, 256, 0, stream>>>(
      hb, wf1b, bf1, ff1, nullptr, nullptr, ROWS, FFDIM, INDIM);
  dim3 gF2(ROWS / 128, INDIM / 128);
  gemm_bf16_wmma<false, true, true><<<gF2, 256, 0, stream>>>(
      ff1, wf2b, bf2, nullptr, (float*)d_out, xo, ROWS, INDIM, FFDIM);
}
